// HBitLinear_26405458936546
// MI455X (gfx1250) — 8-bit matrix operand order repaired and checked by reading, compiled, not run on hardware
//
#include <hip/hip_runtime.h>
#include <hip/hip_bf16.h>

typedef int v8i __attribute__((ext_vector_type(8)));
typedef unsigned int u32x4 __attribute__((ext_vector_type(4)));
typedef int i32x8 __attribute__((ext_vector_type(8)));
typedef int i32x4 __attribute__((ext_vector_type(4)));

#define TOKENS 8192
#define INF    3072
#define PADF   4096
#define NOUT   4096

__global__ __launch_bounds__(256) void k_preprocess(
    const float* __restrict__ x, const float* __restrict__ gamma,
    const float* __restrict__ beta, signed char* __restrict__ Aq,
    float* __restrict__ a_deq)
{
  __shared__ float buf[PADF];
  __shared__ float red1[256];
  __shared__ float red2[256];
  const int tid = threadIdx.x;
  const int token = blockIdx.x;
  const float* xr = x + (size_t)token * INF;

  float s1 = 0.f, s2 = 0.f;
  for (int i = tid; i < INF; i += 256) {
    float v = xr[i];
    buf[i] = v;
    s1 += v; s2 += v * v;
  }
  red1[tid] = s1; red2[tid] = s2;
  __syncthreads();
  for (int off = 128; off > 0; off >>= 1) {
    if (tid < off) { red1[tid] += red1[tid+off]; red2[tid] += red2[tid+off]; }
    __syncthreads();
  }
  const float mu   = red1[0] * (1.f / INF);
  const float var  = red2[0] * (1.f / INF) - mu * mu;
  const float rstd = rsqrtf(var + 1e-6f);
  __syncthreads();

  for (int i = tid; i < INF; i += 256)
    buf[i] = (buf[i] - mu) * rstd * gamma[i] + beta[i];
  for (int i = INF + tid; i < PADF; i += 256)
    buf[i] = 0.f;
  __syncthreads();

  for (int s = 1; s < PADF; s <<= 1) {
    for (int p = tid; p < PADF / 2; p += 256) {
      int i = ((p & ~(s - 1)) << 1) | (p & (s - 1));
      int j = i + s;
      float a = buf[i], b = buf[j];
      buf[i] = a + b;
      buf[j] = a - b;
    }
    __syncthreads();
  }

  float m = 0.f;
  for (int i = tid; i < PADF; i += 256) m = fmaxf(m, fabsf(buf[i]));
  red1[tid] = m;
  __syncthreads();
  for (int off = 128; off > 0; off >>= 1) {
    if (tid < off) red1[tid] = fmaxf(red1[tid], red1[tid+off]);
    __syncthreads();
  }
  const float amax = fmaxf(red1[0] * (1.f / 64.f), 1e-5f);
  const float qs   = (127.f / amax) * (1.f / 64.f);

  signed char* arow = Aq + (size_t)token * PADF + tid * 16;
  int pk[4];
  #pragma unroll
  for (int w4 = 0; w4 < 4; ++w4) {
    int packed = 0;
    #pragma unroll
    for (int e = 0; e < 4; ++e) {
      float q = rintf(buf[tid * 16 + w4 * 4 + e] * qs);
      q = fminf(fmaxf(q, -128.f), 127.f);
      packed |= ((int)q & 0xFF) << (8 * e);
    }
    pk[w4] = packed;
  }
  *(int4*)arow = make_int4(pk[0], pk[1], pk[2], pk[3]);

  if (tid == 0) a_deq[token] = amax * (1.f / 127.f);
}

__global__ __launch_bounds__(256) void k_wabs_partial(
    const float* __restrict__ w, float* __restrict__ partials)
{
  __shared__ float red[256];
  const int tid = threadIdx.x;
  const size_t n4 = (size_t)NOUT * PADF / 4;
  const float4* w4 = (const float4*)w;
  float s = 0.f;
  for (size_t i = (size_t)blockIdx.x * 256 + tid; i < n4; i += (size_t)gridDim.x * 256) {
    float4 v = w4[i];
    s += fabsf(v.x) + fabsf(v.y) + fabsf(v.z) + fabsf(v.w);
  }
  red[tid] = s;
  __syncthreads();
  for (int off = 128; off > 0; off >>= 1) {
    if (tid < off) red[tid] += red[tid + off];
    __syncthreads();
  }
  if (tid == 0) partials[blockIdx.x] = red[0];
}

__global__ __launch_bounds__(256) void k_wfinal(
    const float* __restrict__ partials, float* __restrict__ wscale)
{
  __shared__ float red[256];
  const int tid = threadIdx.x;
  red[tid] = partials[tid] + partials[tid + 256];
  __syncthreads();
  for (int off = 128; off > 0; off >>= 1) {
    if (tid < off) red[tid] += red[tid + off];
    __syncthreads();
  }
  if (tid == 0)
    wscale[0] = fmaxf(red[0] / (float)((size_t)NOUT * PADF), 1e-5f);
}

__global__ __launch_bounds__(256) void k_wquant(
    const float* __restrict__ w, const float* __restrict__ wscale,
    signed char* __restrict__ Wq)
{
  const float inv = 1.f / wscale[0];
  const size_t n4 = (size_t)NOUT * PADF / 4;
  const float4* w4 = (const float4*)w;
  int* q4 = (int*)Wq;
  for (size_t i = (size_t)blockIdx.x * 256 + threadIdx.x; i < n4;
       i += (size_t)gridDim.x * 256) {
    float4 v = w4[i];
    int b0 = (int)fminf(fmaxf(rintf(v.x * inv), -1.f), 1.f);
    int b1 = (int)fminf(fmaxf(rintf(v.y * inv), -1.f), 1.f);
    int b2 = (int)fminf(fmaxf(rintf(v.z * inv), -1.f), 1.f);
    int b3 = (int)fminf(fmaxf(rintf(v.w * inv), -1.f), 1.f);
    q4[i] = (b0 & 0xFF) | ((b1 & 0xFF) << 8) | ((b2 & 0xFF) << 16) | ((b3 & 0xFF) << 24);
  }
}

__device__ __forceinline__ void tdm_load_tile(const void* gsrc, unsigned lds_off,
                                              unsigned rows) {
  const unsigned long long ga = (unsigned long long)(uintptr_t)gsrc;
  u32x4 g0;
  g0[0] = 1u;
  g0[1] = lds_off;
  g0[2] = (unsigned)(ga & 0xFFFFFFFFu);
  g0[3] = (unsigned)((ga >> 32) & 0x01FFFFFFu)
        | (2u << 30);
  i32x8 g1;
  g1[0] = (1 << 20) | (3 << 22) | (3 << 25);
  g1[1] = (int)(4096u << 16);
  g1[2] = (int)(8192u << 16);
  g1[3] = (int)(64u << 16);
  g1[4] = (int)rows;
  g1[5] = 4096;
  g1[6] = 0;
  g1[7] = 0;
  i32x4 z4 = {};
#if __has_include(<hip/amd_detail/amd_gfx1250_TDM.h>)
  i32x8 z8 = {};
  __builtin_amdgcn_tensor_load_to_lds(g0, g1, z4, z4, z8, 0);
#else
  __builtin_amdgcn_tensor_load_to_lds(g0, g1, z4, z4, 0);
#endif
}

__global__ __launch_bounds__(256) void k_gemm_i8(
    const signed char* __restrict__ Aq, const signed char* __restrict__ Wq,
    const float* __restrict__ a_deq, const float* __restrict__ wscale,
    float* __restrict__ out)
{
  __shared__ __align__(16) signed char lsA[2][128 * 80];
  __shared__ __align__(16) signed char lsW[2][128 * 80];

  const int tid  = threadIdx.x;
  const int wave = tid >> 5;
  const int lane = tid & 31;
  const int h    = lane >> 4;
  const int mn   = lane & 15;
  const int N0   = blockIdx.x * 128;
  const int M0   = blockIdx.y * 128;

  const signed char* gA = Aq + (size_t)M0 * PADF;
  const signed char* gW = Wq + (size_t)N0 * PADF;
  const unsigned ldsA[2] = { (unsigned)(size_t)&lsA[0][0],
                             (unsigned)(size_t)&lsA[1][0] };
  const unsigned ldsW[2] = { (unsigned)(size_t)&lsW[0][0],
                             (unsigned)(size_t)&lsW[1][0] };

  v8i acc[8] = {};

  if (wave == 0) {
    tdm_load_tile(gA, ldsA[0], 128);
    tdm_load_tile(gW, ldsW[0], 128);
  }
  __builtin_amdgcn_s_wait_tensorcnt(0);
  __syncthreads();

  for (int k0 = 0; k0 < PADF; k0 += 64) {
    const int cur = (k0 >> 6) & 1;
    if (k0 + 64 < PADF && wave == 0) {
      tdm_load_tile(gA + k0 + 64, ldsA[cur ^ 1], 128);
      tdm_load_tile(gW + k0 + 64, ldsW[cur ^ 1], 128);
    }

    const signed char* pA = lsA[cur] + (wave * 16 + mn) * 80 + h * 8;
    v8i a;
    a[0] = *(const int*)(pA +  0);
    a[1] = *(const int*)(pA +  4);
    a[2] = *(const int*)(pA + 16);
    a[3] = *(const int*)(pA + 20);
    a[4] = *(const int*)(pA + 32);
    a[5] = *(const int*)(pA + 36);
    a[6] = *(const int*)(pA + 48);
    a[7] = *(const int*)(pA + 52);

    v8i b[8];
    #pragma unroll
    for (int t = 0; t < 8; ++t) {
      const signed char* pB = lsW[cur] + (t * 16 + mn) * 80 + h * 8;
      b[t][0] = *(const int*)(pB +  0);
      b[t][1] = *(const int*)(pB +  4);
      b[t][2] = *(const int*)(pB + 16);
      b[t][3] = *(const int*)(pB + 20);
      b[t][4] = *(const int*)(pB + 32);
      b[t][5] = *(const int*)(pB + 36);
      b[t][6] = *(const int*)(pB + 48);
      b[t][7] = *(const int*)(pB + 52);
    }
    #pragma unroll
    for (int t = 0; t < 8; ++t) {
      acc[t] = __builtin_amdgcn_wmma_i32_16x16x64_iu8(
           true, a,  true, b[t], acc[t],
           false,  false);
    }

    __builtin_amdgcn_s_wait_tensorcnt(0);
    __syncthreads();
  }

  const float wsc = wscale[0];
  #pragma unroll
  for (int r = 0; r < 8; ++r) {
    const int m     = r + h * 8;
    const int token = M0 + wave * 16 + m;
    const float sc  = wsc * a_deq[token];
    float* orow = out + (size_t)token * NOUT + N0 + mn;
    #pragma unroll
    for (int t = 0; t < 8; ++t)
      orow[t * 16] = (float)acc[t][r] * sc;
  }
}

extern "C" void kernel_launch(void* const* d_in, const int* in_sizes, int n_in,
                              void* d_out, int out_size, void* d_ws, size_t ws_size,
                              hipStream_t stream) {
  const float* x      = (const float*)d_in[0];
  const float* gamma  = (const float*)d_in[1];
  const float* beta   = (const float*)d_in[2];
  const float* weight = (const float*)d_in[3];
  float* out = (float*)d_out;

  char* ws = (char*)d_ws;
  signed char* Aq    = (signed char*)ws;
  signed char* Wq    = (signed char*)(ws + 33554432);
  float* a_deq       = (float*)(ws + 50331648);
  float* partials    = (float*)(ws + 50331648 + 32768);
  float* wscale      = (float*)(ws + 50331648 + 32768 + 2048);

  k_preprocess  <<<TOKENS,        256, 0, stream>>>(x, gamma, beta, Aq, a_deq);
  k_wabs_partial<<<512,           256, 0, stream>>>(weight, partials);
  k_wfinal      <<<1,             256, 0, stream>>>(partials, wscale);
  k_wquant      <<<4096,          256, 0, stream>>>(weight, wscale, Wq);
  k_gemm_i8     <<<dim3(32, 64),  256, 0, stream>>>(Aq, Wq, a_deq, wscale, out);
}
